// DiTTranslator_6777458393116
// MI455X (gfx1250) — matrix operand order repaired and checked by reading, not compiled, not run on hardware
//
#include <hip/hip_runtime.h>
#include <hip/hip_bf16.h>
#include <math.h>
#include <stdint.h>

typedef __bf16 bf16_t;
typedef __bf16 v16bf __attribute__((ext_vector_type(16)));
typedef float  v8f   __attribute__((ext_vector_type(8)));

#define BM 128
#define BN 64
#define BK 32
#define BKF 36   // fp32 LDS row pad: 144B stride keeps 16B alignment, skews banks
#define BKP 40   // bf16 LDS row pad: 80B stride keeps 16B alignment

#ifndef USE_ASYNC_LDS
#define USE_ASYNC_LDS 1
#endif

// ---------- helpers ----------
// pack two fp32 -> two bf16 (round-to-nearest-even) in one u32
__device__ inline unsigned pkbf(float x, float y) {
  unsigned a = __builtin_bit_cast(unsigned, x);
  unsigned b = __builtin_bit_cast(unsigned, y);
  a = (a + 0x7FFFu + ((a >> 16) & 1u)) >> 16;
  b = (b + 0x7FFFu + ((b >> 16) & 1u)) & 0xFFFF0000u;
  return a | b;
}

__device__ inline float wred_sum(float v) {
#pragma unroll
  for (int o = 16; o > 0; o >>= 1) v += __shfl_xor(v, o, 32);
  return v;
}
__device__ inline float wred_max(float v) {
#pragma unroll
  for (int o = 16; o > 0; o >>= 1) v = fmaxf(v, __shfl_xor(v, o, 32));
  return v;
}

#if USE_ASYNC_LDS
// LDS byte address of a __shared__ object == low 32 bits of its flat address.
__device__ inline unsigned lds_addr32(const void* p) {
  return (unsigned)(uintptr_t)p;
}
// CDNA5 async global->LDS copies (tracked by ASYNCcnt). Each lane copies its own
// 16B / 4B chunk: lane supplies both the global source and the LDS destination.
__device__ inline void async_b128(const void* lds, const float* g) {
  asm volatile("global_load_async_to_lds_b128 %0, %1, off"
               :: "v"(lds_addr32(lds)), "v"((unsigned long long)(uintptr_t)g)
               : "memory");
}
__device__ inline void async_b32(const void* lds, const float* g) {
  asm volatile("global_load_async_to_lds_b32 %0, %1, off"
               :: "v"(lds_addr32(lds)), "v"((unsigned long long)(uintptr_t)g)
               : "memory");
}
__device__ inline void wait_async() {
  asm volatile("s_wait_asynccnt 0x0" ::: "memory");
}
#endif

// ---------- general batched GEMM with bf16 WMMA ----------
// C[z] = epilogue( alpha * A[z] @ op(B[z]) + bias )
// mode: 0 = linear(+bias), 1 = gelu(+bias), 2 = res + t, 3 = res + tanh(*gate)*t
// z decomposed as zb = z / nz2, z2 = z % nz2 (strides sX1/sX2) for per-(batch,head) GEMMs.
__global__ __launch_bounds__(256) void gemm_bf16_kernel(
    const float* __restrict__ A, int lda, long sA1, long sA2,
    const float* __restrict__ B, int ldb, long sB1, long sB2, int transB,
    float* __restrict__ C, int ldc, long sC1, long sC2,
    const float* __restrict__ bias,
    const float* __restrict__ res, long sR1, long sR2,
    const float* __restrict__ gate,
    int M, int N, int K, float alpha, int mode, int nz2)
{
  __shared__ float  Af[BM][BKF];   // fp32 tile (async landing zone)
  __shared__ float  Bf[BN][BKF];   // fp32 tile, N-major
  __shared__ bf16_t Ab[BM][BKP];   // bf16 tile for fragment loads
  __shared__ bf16_t Bb[BN][BKP];

  const int z = blockIdx.z, zb = z / nz2, z2 = z - zb * nz2;
  A += zb * sA1 + z2 * sA2;
  B += zb * sB1 + z2 * sB2;
  C += zb * sC1 + z2 * sC2;
  if (res) res += zb * sR1 + z2 * sR2;

  const int m_blk = blockIdx.y * BM, n_blk = blockIdx.x * BN;
  const int tid = threadIdx.x, lane = tid & 31, w = tid >> 5;
  const int wm = (w >> 1) << 5;  // 0,32,64,96
  const int wn = (w & 1) << 5;   // 0,32

  // Block-uniform fast path: full tile, K multiple of BK (true for all big GEMMs here)
  const bool full = (m_blk + BM <= M) && (n_blk + BN <= N) && ((K & (BK - 1)) == 0);

  v8f acc[2][2] = {};

  for (int k0 = 0; k0 < K; k0 += BK) {
    if (full) {
#if USE_ASYNC_LDS
      // ---- async stage A (BM x BK fp32): 1024 x 16B chunks, no VGPR round-trip ----
#pragma unroll
      for (int p = 0; p < 4; ++p) {
        int i = tid + (p << 8);
        int r = i >> 3, c4 = (i & 7) << 2;
        async_b128(&Af[r][c4], A + (long)(m_blk + r) * lda + k0 + c4);
      }
      if (transB) {
#pragma unroll
        for (int p = 0; p < 2; ++p) {
          int i = tid + (p << 8);
          int n = i >> 3, c4 = (i & 7) << 2;
          async_b128(&Bf[n][c4], B + (long)(n_blk + n) * ldb + k0 + c4);
        }
      } else {
        // transpose-stage [K,N] -> Bf[n][k] with per-lane b32 scatter
#pragma unroll
        for (int p = 0; p < 8; ++p) {
          int i = tid + (p << 8);
          int k = i >> 6, n = i & 63;
          async_b32(&Bf[n][k], B + (long)(k0 + k) * ldb + n_blk + n);
        }
      }
      wait_async();
#else
#pragma unroll
      for (int p = 0; p < 4; ++p) {
        int i = tid + (p << 8);
        int r = i >> 3, c4 = (i & 7) << 2;
        float4 v = *(const float4*)(A + (long)(m_blk + r) * lda + k0 + c4);
        float* d = &Af[r][c4];
        d[0] = v.x; d[1] = v.y; d[2] = v.z; d[3] = v.w;
      }
      if (transB) {
#pragma unroll
        for (int p = 0; p < 2; ++p) {
          int i = tid + (p << 8);
          int n = i >> 3, c4 = (i & 7) << 2;
          float4 v = *(const float4*)(B + (long)(n_blk + n) * ldb + k0 + c4);
          float* d = &Bf[n][c4];
          d[0] = v.x; d[1] = v.y; d[2] = v.z; d[3] = v.w;
        }
      } else {
#pragma unroll
        for (int p = 0; p < 2; ++p) {
          int i = tid + (p << 8);
          int k = i >> 4, nb = (i & 15) << 2;
          float4 v = *(const float4*)(B + (long)(k0 + k) * ldb + n_blk + nb);
          Bf[nb][k] = v.x; Bf[nb + 1][k] = v.y; Bf[nb + 2][k] = v.z; Bf[nb + 3][k] = v.w;
        }
      }
#endif
    } else {
      // ---- guarded edge path (small M tiles: M=64 attn batches, M=2 temb) ----
#pragma unroll
      for (int p = 0; p < 4; ++p) {
        int i = tid + (p << 8);
        int r = i >> 3, c4 = (i & 7) << 2;
        int gm = m_blk + r, gk = k0 + c4;
        float4 v = make_float4(0.f, 0.f, 0.f, 0.f);
        if (gm < M) {
          const float* ap = A + (long)gm * lda;
          if (gk + 3 < K) v = *(const float4*)(ap + gk);
          else {
            if (gk     < K) v.x = ap[gk];
            if (gk + 1 < K) v.y = ap[gk + 1];
            if (gk + 2 < K) v.z = ap[gk + 2];
          }
        }
        float* d = &Af[r][c4];
        d[0] = v.x; d[1] = v.y; d[2] = v.z; d[3] = v.w;
      }
      if (transB) {
#pragma unroll
        for (int p = 0; p < 2; ++p) {
          int i = tid + (p << 8);
          int n = i >> 3, c4 = (i & 7) << 2;
          int gn = n_blk + n, gk = k0 + c4;
          float4 v = make_float4(0.f, 0.f, 0.f, 0.f);
          if (gn < N) {
            const float* bp = B + (long)gn * ldb;
            if (gk + 3 < K) v = *(const float4*)(bp + gk);
            else {
              if (gk     < K) v.x = bp[gk];
              if (gk + 1 < K) v.y = bp[gk + 1];
              if (gk + 2 < K) v.z = bp[gk + 2];
            }
          }
          float* d = &Bf[n][c4];
          d[0] = v.x; d[1] = v.y; d[2] = v.z; d[3] = v.w;
        }
      } else {
#pragma unroll
        for (int p = 0; p < 2; ++p) {
          int i = tid + (p << 8);
          int k = i >> 4, nb = (i & 15) << 2;
          int gk = k0 + k, gn = n_blk + nb;
          float4 v = make_float4(0.f, 0.f, 0.f, 0.f);
          if (gk < K) {
            const float* bp = B + (long)gk * ldb;
            if (gn + 3 < N) v = *(const float4*)(bp + gn);
            else {
              if (gn     < N) v.x = bp[gn];
              if (gn + 1 < N) v.y = bp[gn + 1];
              if (gn + 2 < N) v.z = bp[gn + 2];
            }
          }
          Bf[nb][k] = v.x; Bf[nb + 1][k] = v.y; Bf[nb + 2][k] = v.z; Bf[nb + 3][k] = v.w;
        }
      }
    }
    // prefetch next A tile rows into L2 (gfx1250 global_prefetch_b8)
    if (k0 + BK < K && tid < BM) {
      int gm = m_blk + tid;
      if (gm < M) __builtin_prefetch((const void*)(A + (long)gm * lda + k0 + BK), 0, 1);
    }
    __syncthreads();   // fp32 tiles visible

    // ---- LDS->LDS conversion pass: fp32 tile -> packed bf16 tile ----
#pragma unroll
    for (int p = 0; p < 4; ++p) {            // A: 1024 float4 chunks
      int i = tid + (p << 8);
      int r = i >> 3, c4 = (i & 7) << 2;
      const float* s = &Af[r][c4];
      unsigned lo = pkbf(s[0], s[1]);
      unsigned hi = pkbf(s[2], s[3]);
      unsigned* d = (unsigned*)&Ab[r][c4];
      d[0] = lo; d[1] = hi;
    }
#pragma unroll
    for (int p = 0; p < 2; ++p) {            // B: 512 float4 chunks
      int i = tid + (p << 8);
      int n = i >> 3, c4 = (i & 7) << 2;
      const float* s = &Bf[n][c4];
      unsigned lo = pkbf(s[0], s[1]);
      unsigned hi = pkbf(s[2], s[3]);
      unsigned* d = (unsigned*)&Bb[n][c4];
      d[0] = lo; d[1] = hi;
    }
    __syncthreads();   // bf16 tiles visible

    // ---- fragments are pure contiguous ds_load_b128s, then 4 WMMAs ----
    {
      const int kb_a = (lane >> 4) << 3;   // A: lanes 0-15 -> K {0..7,16..23}; 16-31 -> {8..15,24..31}
      const int kb_b = (lane >> 4) << 3;   // B: lanes 0-15 -> K {0..7,16..23}; 16-31 -> {8..15,24..31}
      const int ml = lane & 15;
      v16bf af[2], bfr[2];
#pragma unroll
      for (int t = 0; t < 2; ++t) {
        const bf16_t* pa = &Ab[wm + (t << 4) + ml][kb_a];
#pragma unroll
        for (int j = 0; j < 8; ++j) { af[t][j] = pa[j]; af[t][8 + j] = pa[16 + j]; }
        const bf16_t* pb = &Bb[wn + (t << 4) + ml][kb_b];
#pragma unroll
        for (int j = 0; j < 8; ++j) { bfr[t][j] = pb[j]; bfr[t][8 + j] = pb[16 + j]; }
      }
#pragma unroll
      for (int tm = 0; tm < 2; ++tm)
#pragma unroll
        for (int tn = 0; tn < 2; ++tn)
          acc[tm][tn] = __builtin_amdgcn_wmma_f32_16x16x32_bf16(
              false, af[tm], false, bfr[tn], (short)0, acc[tm][tn], false, false);
    }
    __syncthreads();   // safe to overwrite tiles
  }

  // ---- epilogue (C/D layout: lane&15 = n, VGPR r -> m = r + (lane>>4)*8) ----
  const float gv = (mode == 3 && gate) ? tanhf(*gate) : 1.0f;
  const int ml = lane & 15, mh = (lane >> 4) << 3;
#pragma unroll
  for (int tm = 0; tm < 2; ++tm) {
#pragma unroll
    for (int tn = 0; tn < 2; ++tn) {
      int gn = n_blk + wn + (tn << 4) + ml;
      if (gn >= N) continue;
      float bv = bias ? bias[gn] : 0.f;
#pragma unroll
      for (int r = 0; r < 8; ++r) {
        int gm = m_blk + wm + (tm << 4) + mh + r;
        if (gm >= M) continue;
        float t = acc[tm][tn][r] * alpha + bv;
        if (mode == 1)      t = 0.5f * t * (1.0f + erff(t * 0.70710678118654752f));
        else if (mode == 2) t = res[(long)gm * ldc + gn] + t;
        else if (mode == 3) t = res[(long)gm * ldc + gn] + gv * t;
        C[(long)gm * ldc + gn] = t;
      }
    }
  }
}

// ---------- layernorm (one 256-thread block per row) ----------
__global__ __launch_bounds__(256) void ln_kernel(
    const float* __restrict__ x, const float* __restrict__ g,
    const float* __restrict__ b, float* __restrict__ y, int width)
{
  long row = blockIdx.x;
  const float* xr = x + row * (long)width;
  float* yr = y + row * (long)width;
  __shared__ float sh[2][8];
  float s = 0.f, s2 = 0.f;
  for (int i = threadIdx.x; i < width; i += 256) { float v = xr[i]; s += v; s2 += v * v; }
  s = wred_sum(s); s2 = wred_sum(s2);
  int w = threadIdx.x >> 5, lane = threadIdx.x & 31;
  if (lane == 0) { sh[0][w] = s; sh[1][w] = s2; }
  __syncthreads();
  s = 0.f; s2 = 0.f;
#pragma unroll
  for (int i = 0; i < 8; ++i) { s += sh[0][i]; s2 += sh[1][i]; }
  float mean = s / width;
  float var  = s2 / width - mean * mean;
  float rstd = rsqrtf(var + 1e-5f);
  for (int i = threadIdx.x; i < width; i += 256)
    yr[i] = (xr[i] - mean) * rstd * g[i] + b[i];
}

// ---------- row softmax (scale pre-applied by GEMM alpha); optional int mask ----------
__global__ __launch_bounds__(256) void softmax_kernel(
    float* __restrict__ S, int n, int rows_per_batch,
    const int* __restrict__ mask, int mask_ld)
{
  long row = blockIdx.x;
  float* sr = S + row * (long)n;
  const int* mr = mask ? mask + (row / rows_per_batch) * (long)mask_ld : nullptr;
  __shared__ float sh[8];
  int w = threadIdx.x >> 5, lane = threadIdx.x & 31;
  float mx = -3.0e38f;
  for (int i = threadIdx.x; i < n; i += 256) {
    float v = sr[i];
    if (mr && mr[i] == 0) v = -10000.0f;
    mx = fmaxf(mx, v);
  }
  mx = wred_max(mx);
  if (lane == 0) sh[w] = mx;
  __syncthreads();
  mx = sh[0];
#pragma unroll
  for (int i = 1; i < 8; ++i) mx = fmaxf(mx, sh[i]);
  __syncthreads();
  float sum = 0.f;
  for (int i = threadIdx.x; i < n; i += 256) {
    float v = sr[i];
    if (mr && mr[i] == 0) v = -10000.0f;
    float e = __expf(v - mx);
    sr[i] = e;
    sum += e;
  }
  sum = wred_sum(sum);
  if (lane == 0) sh[w] = sum;
  __syncthreads();
  sum = 0.f;
#pragma unroll
  for (int i = 0; i < 8; ++i) sum += sh[i];
  float inv = 1.0f / sum;
  for (int i = threadIdx.x; i < n; i += 256) sr[i] *= inv;
}

// ---------- timestep embedding: te0[b, 0:512]=cos(t*f), [512:1024]=sin(t*f) ----------
__global__ void tstep_kernel(const float* __restrict__ t, float* __restrict__ te0) {
  int b = blockIdx.x;
  int i = blockIdx.y * 256 + threadIdx.x;   // 0..1023
  int j = i & 511;
  float f = __expf(-9.210340371976184f * (float)j * (1.0f / 512.0f));
  float a = t[b] * f;
  te0[b * 1024 + i] = (i < 512) ? __cosf(a) : __sinf(a);
}

// ---------- x[b,t,:] += te[b,:] ----------
__global__ void addb_kernel(float* __restrict__ x, const float* __restrict__ te,
                            int T, int C, long total) {
  long idx = (long)blockIdx.x * 256 + threadIdx.x;
  if (idx >= total) return;
  long b = idx / ((long)T * C);
  int c = (int)(idx % C);
  x[idx] += te[b * C + c];
}

// ---------- broadcast query [1,S,C] -> [B,S,C] ----------
__global__ void bquery_kernel(const float* __restrict__ q, float* __restrict__ x,
                              long per_batch, long total) {
  long idx = (long)blockIdx.x * 256 + threadIdx.x;
  if (idx < total) x[idx] = q[idx % per_batch];
}

// ================= host orchestration =================
enum {
  IN_SRC_H = 0, IN_SRC_MASK, IN_X_T, IN_X_MASK, IN_T,
  R_QUERY, R_SRC_PROJ_W, R_SRC_PROJ_B, R_LNQ_G, R_LNQ_B, R_LNKV_G, R_LNKV_B,
  R_SA_WQKV, R_SA_BQKV, R_SA_WO, R_SA_BO,
  R_CQ_W, R_CQ_B, R_CK_W, R_CK_B, R_CV_W, R_CV_B, R_CO_W, R_CO_B,
  R_FFN_W1, R_FFN_B1, R_FFN_W2, R_FFN_B2, R_GATE,
  D_LN1_G, D_LN1_B, D_SA_WQKV, D_SA_BQKV, D_SA_WO, D_SA_BO,
  D_LN2X_G, D_LN2X_B, D_LN2C_G, D_LN2C_B,
  D_CQ_W, D_CQ_B, D_CK_W, D_CK_B, D_CV_W, D_CV_B, D_CO_W, D_CO_B,
  D_GATE, D_LN3_G, D_LN3_B, D_FFN_W1, D_FFN_B1, D_FFN_W2, D_FFN_B2,
  T_DOWN_W, T_DOWN_B, T_TEMB_W1, T_TEMB_B1, T_TEMB_W2, T_TEMB_B2,
  T_LNOUT_G, T_LNOUT_B, T_OUT_W, T_OUT_B
};

static void launch_gemm(hipStream_t st,
    const float* A, int lda, long sA1, long sA2,
    const float* B, int ldb, long sB1, long sB2, int transB,
    float* C, int ldc, long sC1, long sC2,
    const float* bias, const float* res, long sR1, long sR2,
    const float* gate, int M, int N, int K, float alpha, int mode,
    int batch, int nz2)
{
  dim3 grid((N + BN - 1) / BN, (M + BM - 1) / BM, batch);
  gemm_bf16_kernel<<<grid, 256, 0, st>>>(A, lda, sA1, sA2, B, ldb, sB1, sB2, transB,
      C, ldc, sC1, sC2, bias, res, sR1, sR2, gate, M, N, K, alpha, mode, nz2);
}

extern "C" void kernel_launch(void* const* d_in, const int* in_sizes, int n_in,
                              void* d_out, int out_size, void* d_ws, size_t ws_size,
                              hipStream_t stream)
{
  (void)in_sizes; (void)n_in; (void)out_size; (void)ws_size;
  auto F = [&](int i) { return (const float*)d_in[i]; };
  const int* src_mask = (const int*)d_in[IN_SRC_MASK];

  float* ws = (float*)d_ws;
  const long SZ2M = 2l * 1024 * 1024;
  long o = 0;
  float* XN  = ws + o; o += SZ2M;               // LN scratch (up to 2048x1024)
  float* QKV = ws + o; o += 2l * 1024 * 3072;   // qkv projections
  float* SC  = ws + o; o += 32l * 1024 * 1024;  // MHA scores (L2-resident, 134MB)
  float* ATT = ws + o; o += SZ2M;               // attention output (pre-Wo)
  float* KV  = ws + o; o += SZ2M;               // resampler kv
  float* KVN = ws + o; o += SZ2M;               // LN(kv) / LN(cond)
  float* QB  = ws + o; o += SZ2M;               // cross q
  float* KB  = ws + o; o += SZ2M;               // cross k
  float* VB  = ws + o; o += SZ2M;               // cross v
  float* SC2 = ws + o; o += 2l * 1024 * 64;     // cross scores
  float* CR  = ws + o; o += SZ2M;               // cross context
  float* H1  = ws + o; o += 2l * 1024 * 4096;   // FFN hidden
  float* X   = ws + o; o += SZ2M;               // DiT activations
  float* RX  = ws + o; o += 2l * 64 * 1024;     // resampler latents / cond
  float* RXN = ws + o; o += 2l * 64 * 1024;     // resampler LN scratch
  float* TE0 = ws + o; o += 2l * 1024;
  float* TE1 = ws + o; o += 2l * 4096;
  float* TE  = ws + o; o += 2l * 1024;

  // ===================== Resampler =====================
  launch_gemm(stream, F(IN_SRC_H), 4096, 0, 0, F(R_SRC_PROJ_W), 1024, 0, 0, 0,
              KV, 1024, 0, 0, F(R_SRC_PROJ_B), nullptr, 0, 0, nullptr,
              2048, 1024, 4096, 1.f, 0, 1, 1);
  { long total = 2l * 64 * 1024;
    bquery_kernel<<<(total + 255) / 256, 256, 0, stream>>>(F(R_QUERY), RX, 64l * 1024, total); }

  for (int i = 0; i < 2; ++i) {
    const float* lnq_g  = F(R_LNQ_G)  + i * 1024;
    const float* lnq_b  = F(R_LNQ_B)  + i * 1024;
    const float* lnkv_g = F(R_LNKV_G) + i * 1024;
    const float* lnkv_b = F(R_LNKV_B) + i * 1024;
    const long W1 = (long)i * 1024 * 1024, W3 = (long)i * 1024 * 3072, W4 = (long)i * 1024 * 4096;

    // --- self-attention (heads=8, d=128, T=64) ---
    ln_kernel<<<128, 256, 0, stream>>>(RX, lnq_g, lnq_b, RXN, 1024);
    launch_gemm(stream, RXN, 1024, 0, 0, F(R_SA_WQKV) + W3, 3072, 0, 0, 0,
                QKV, 3072, 0, 0, F(R_SA_BQKV) + i * 3072, nullptr, 0, 0, nullptr,
                128, 3072, 1024, 1.f, 0, 1, 1);
    launch_gemm(stream, QKV, 3072, 64l * 3072, 128, QKV + 1024, 3072, 64l * 3072, 128, 1,
                SC, 64, 8l * 64 * 64, 64l * 64, nullptr, nullptr, 0, 0, nullptr,
                64, 64, 128, 0.08838834764831845f, 0, 16, 8);
    softmax_kernel<<<16 * 64, 256, 0, stream>>>(SC, 64, 64, nullptr, 0);
    launch_gemm(stream, SC, 64, 8l * 64 * 64, 64l * 64, QKV + 2048, 3072, 64l * 3072, 128, 0,
                ATT, 1024, 64l * 1024, 128, nullptr, nullptr, 0, 0, nullptr,
                64, 128, 64, 1.f, 0, 16, 8);
    launch_gemm(stream, ATT, 1024, 0, 0, F(R_SA_WO) + W1, 1024, 0, 0, 0,
                RX, 1024, 0, 0, F(R_SA_BO) + i * 1024, RX, 0, 0, nullptr,
                128, 1024, 1024, 1.f, 2, 1, 1);

    // --- gated cross-attention (single head, d=1024) ---
    ln_kernel<<<128, 256, 0, stream>>>(RX, lnq_g, lnq_b, RXN, 1024);
    ln_kernel<<<2048, 256, 0, stream>>>(KV, lnkv_g, lnkv_b, KVN, 1024);
    launch_gemm(stream, RXN, 1024, 0, 0, F(R_CQ_W) + W1, 1024, 0, 0, 0,
                QB, 1024, 0, 0, F(R_CQ_B) + i * 1024, nullptr, 0, 0, nullptr,
                128, 1024, 1024, 1.f, 0, 1, 1);
    launch_gemm(stream, KVN, 1024, 0, 0, F(R_CK_W) + W1, 1024, 0, 0, 0,
                KB, 1024, 0, 0, F(R_CK_B) + i * 1024, nullptr, 0, 0, nullptr,
                2048, 1024, 1024, 1.f, 0, 1, 1);
    launch_gemm(stream, KVN, 1024, 0, 0, F(R_CV_W) + W1, 1024, 0, 0, 0,
                VB, 1024, 0, 0, F(R_CV_B) + i * 1024, nullptr, 0, 0, nullptr,
                2048, 1024, 1024, 1.f, 0, 1, 1);
    launch_gemm(stream, QB, 1024, 64l * 1024, 0, KB, 1024, 1024l * 1024, 0, 1,
                SC2, 1024, 64l * 1024, 0, nullptr, nullptr, 0, 0, nullptr,
                64, 1024, 1024, 0.03125f, 0, 2, 1);
    softmax_kernel<<<128, 256, 0, stream>>>(SC2, 1024, 64, src_mask, 1024);
    launch_gemm(stream, SC2, 1024, 64l * 1024, 0, VB, 1024, 1024l * 1024, 0, 0,
                CR, 1024, 64l * 1024, 0, nullptr, nullptr, 0, 0, nullptr,
                64, 1024, 1024, 1.f, 0, 2, 1);
    launch_gemm(stream, CR, 1024, 0, 0, F(R_CO_W) + W1, 1024, 0, 0, 0,
                RX, 1024, 0, 0, F(R_CO_B) + i * 1024, RX, 0, 0, F(R_GATE),
                128, 1024, 1024, 1.f, 3, 1, 1);

    // --- FFN ---
    ln_kernel<<<128, 256, 0, stream>>>(RX, lnq_g, lnq_b, RXN, 1024);
    launch_gemm(stream, RXN, 1024, 0, 0, F(R_FFN_W1) + W4, 4096, 0, 0, 0,
                H1, 4096, 0, 0, F(R_FFN_B1) + i * 4096, nullptr, 0, 0, nullptr,
                128, 4096, 1024, 1.f, 1, 1, 1);
    launch_gemm(stream, H1, 4096, 0, 0, F(R_FFN_W2) + (long)i * 4096 * 1024, 1024, 0, 0, 0,
                RX, 1024, 0, 0, F(R_FFN_B2) + i * 1024, RX, 0, 0, nullptr,
                128, 1024, 4096, 1.f, 2, 1, 1);
  }
  // RX is now `cond` [2,64,1024]

  // ===================== Top: down-proj + time embedding =====================
  launch_gemm(stream, F(IN_X_T), 4096, 0, 0, F(T_DOWN_W), 1024, 0, 0, 0,
              X, 1024, 0, 0, F(T_DOWN_B), nullptr, 0, 0, nullptr,
              2048, 1024, 4096, 1.f, 0, 1, 1);
  tstep_kernel<<<dim3(2, 4), 256, 0, stream>>>(F(IN_T), TE0);
  launch_gemm(stream, TE0, 1024, 0, 0, F(T_TEMB_W1), 4096, 0, 0, 0,
              TE1, 4096, 0, 0, F(T_TEMB_B1), nullptr, 0, 0, nullptr,
              2, 4096, 1024, 1.f, 1, 1, 1);
  launch_gemm(stream, TE1, 4096, 0, 0, F(T_TEMB_W2), 1024, 0, 0, 0,
              TE, 1024, 0, 0, F(T_TEMB_B2), nullptr, 0, 0, nullptr,
              2, 1024, 4096, 1.f, 0, 1, 1);
  { long total = 2l * 1024 * 1024;
    addb_kernel<<<(total + 255) / 256, 256, 0, stream>>>(X, TE, 1024, 1024, total); }

  // ===================== DiT layers =====================
  for (int i = 0; i < 6; ++i) {
    const long W1 = (long)i * 1024 * 1024, W3 = (long)i * 1024 * 3072,
               W4 = (long)i * 1024 * 4096, W4b = (long)i * 4096 * 1024;

    // --- self-attention (heads=16, d=64, T=1024); scores stay in L2 ---
    ln_kernel<<<2048, 256, 0, stream>>>(X, F(D_LN1_G) + i * 1024, F(D_LN1_B) + i * 1024, XN, 1024);
    launch_gemm(stream, XN, 1024, 0, 0, F(D_SA_WQKV) + W3, 3072, 0, 0, 0,
                QKV, 3072, 0, 0, F(D_SA_BQKV) + i * 3072, nullptr, 0, 0, nullptr,
                2048, 3072, 1024, 1.f, 0, 1, 1);
    launch_gemm(stream, QKV, 3072, 1024l * 3072, 64, QKV + 1024, 3072, 1024l * 3072, 64, 1,
                SC, 1024, 16l * 1024 * 1024, 1024l * 1024, nullptr, nullptr, 0, 0, nullptr,
                1024, 1024, 64, 0.125f, 0, 32, 16);
    softmax_kernel<<<32 * 1024, 256, 0, stream>>>(SC, 1024, 1024, nullptr, 0);
    launch_gemm(stream, SC, 1024, 16l * 1024 * 1024, 1024l * 1024, QKV + 2048, 3072, 1024l * 3072, 64, 0,
                ATT, 1024, 1024l * 1024, 64, nullptr, nullptr, 0, 0, nullptr,
                1024, 64, 1024, 1.f, 0, 32, 16);
    launch_gemm(stream, ATT, 1024, 0, 0, F(D_SA_WO) + W1, 1024, 0, 0, 0,
                X, 1024, 0, 0, F(D_SA_BO) + i * 1024, X, 0, 0, nullptr,
                2048, 1024, 1024, 1.f, 2, 1, 1);

    // --- gated cross-attention against cond (single head, d=1024, Tk=64) ---
    ln_kernel<<<2048, 256, 0, stream>>>(X, F(D_LN2X_G) + i * 1024, F(D_LN2X_B) + i * 1024, XN, 1024);
    launch_gemm(stream, XN, 1024, 0, 0, F(D_CQ_W) + W1, 1024, 0, 0, 0,
                QB, 1024, 0, 0, F(D_CQ_B) + i * 1024, nullptr, 0, 0, nullptr,
                2048, 1024, 1024, 1.f, 0, 1, 1);
    ln_kernel<<<128, 256, 0, stream>>>(RX, F(D_LN2C_G) + i * 1024, F(D_LN2C_B) + i * 1024, KVN, 1024);
    launch_gemm(stream, KVN, 1024, 0, 0, F(D_CK_W) + W1, 1024, 0, 0, 0,
                KB, 1024, 0, 0, F(D_CK_B) + i * 1024, nullptr, 0, 0, nullptr,
                128, 1024, 1024, 1.f, 0, 1, 1);
    launch_gemm(stream, KVN, 1024, 0, 0, F(D_CV_W) + W1, 1024, 0, 0, 0,
                VB, 1024, 0, 0, F(D_CV_B) + i * 1024, nullptr, 0, 0, nullptr,
                128, 1024, 1024, 1.f, 0, 1, 1);
    launch_gemm(stream, QB, 1024, 1024l * 1024, 0, KB, 1024, 64l * 1024, 0, 1,
                SC2, 64, 1024l * 64, 0, nullptr, nullptr, 0, 0, nullptr,
                1024, 64, 1024, 0.03125f, 0, 2, 1);
    softmax_kernel<<<2048, 256, 0, stream>>>(SC2, 64, 1024, nullptr, 0);
    launch_gemm(stream, SC2, 64, 1024l * 64, 0, VB, 1024, 64l * 1024, 0, 0,
                CR, 1024, 1024l * 1024, 0, nullptr, nullptr, 0, 0, nullptr,
                1024, 1024, 64, 1.f, 0, 2, 1);
    launch_gemm(stream, CR, 1024, 0, 0, F(D_CO_W) + W1, 1024, 0, 0, 0,
                X, 1024, 0, 0, F(D_CO_B) + i * 1024, X, 0, 0, F(D_GATE) + i,
                2048, 1024, 1024, 1.f, 3, 1, 1);

    // --- FFN ---
    ln_kernel<<<2048, 256, 0, stream>>>(X, F(D_LN3_G) + i * 1024, F(D_LN3_B) + i * 1024, XN, 1024);
    launch_gemm(stream, XN, 1024, 0, 0, F(D_FFN_W1) + W4, 4096, 0, 0, 0,
                H1, 4096, 0, 0, F(D_FFN_B1) + i * 4096, nullptr, 0, 0, nullptr,
                2048, 4096, 1024, 1.f, 1, 1, 1);
    launch_gemm(stream, H1, 4096, 0, 0, F(D_FFN_W2) + W4b, 1024, 0, 0, 0,
                X, 1024, 0, 0, F(D_FFN_B2) + i * 1024, X, 0, 0, nullptr,
                2048, 1024, 4096, 1.f, 2, 1, 1);
  }

  // ===================== Output head =====================
  ln_kernel<<<2048, 256, 0, stream>>>(X, F(T_LNOUT_G), F(T_LNOUT_B), XN, 1024);
  launch_gemm(stream, XN, 1024, 0, 0, F(T_OUT_W), 1024, 0, 0, 0,
              (float*)d_out, 1024, 0, 0, F(T_OUT_B), nullptr, 0, 0, nullptr,
              2048, 1024, 1024, 1.f, 0, 1, 1);
}
